// CrossLinearAttention_31430570672883
// MI455X (gfx1250) — matrix operand order repaired and checked by reading, not compiled, not run on hardware
//
#include <hip/hip_runtime.h>
#include <hip/hip_bf16.h>

#define B_ 8
#define C_ 128
#define N_ 65536      // 256*256
#define H_ 4
#define D_ 32
#define SCALE_ 0.17677669529663687f   // 32^-0.5
#define EPS_ 1e-5f

typedef __bf16 bf16;
typedef __attribute__((ext_vector_type(16))) __bf16 v16bf;
typedef __attribute__((ext_vector_type(8)))  float  v8f;

// ---------------- WMMA fragment helpers (16-bit, contiguous k order, wave32) -----
// A (16x32, MxK): lanes 0-15 / 16-31 both hold M=lane&15.
//   elems 0..15 = K { half*16 + 0..15 }
__device__ __forceinline__ v16bf load_frag_a(const bf16* __restrict__ base,
                                             int stride, int lane) {
  const int row   = lane & 15;
  const int khalf = (lane >> 4) << 4;           // 0 or 16
  const bf16* p = base + (size_t)row * stride + khalf;
  v16bf a;
#pragma unroll
  for (int j = 0; j < 8; ++j) a[j] = p[j];
#pragma unroll
  for (int j = 0; j < 8; ++j) a[8 + j] = p[8 + j];
  return a;
}

// B (32x16, KxN): lanes 0-15 hold K=0..15, lanes 16-31 hold K=16..31, N=lane&15.
// Source stored [N][K] with K contiguous (stride = row pitch in elems).
__device__ __forceinline__ v16bf load_frag_b(const bf16* __restrict__ base,
                                             int stride, int lane) {
  const int n    = lane & 15;
  const int koff = (lane >> 4) << 4;            // 0 or 16
  const bf16* p = base + (size_t)n * stride + koff;
  v16bf b;
#pragma unroll
  for (int j = 0; j < 16; ++j) b[j] = p[j];
  return b;
}

__device__ __forceinline__ v8f wmma_bf16(v16bf a, v16bf b, v8f c) {
  return __builtin_amdgcn_wmma_f32_16x16x32_bf16(
      /*neg_a=*/false, a, /*neg_b=*/false, b,
      /*c_mod=*/(short)0, c, /*reuse_a=*/false, /*reuse_b=*/false);
}

// ---------------- K0: pack weights to bf16 (stay resident in 192MB L2) -----
__global__ void k_pack(const float* __restrict__ Wq, const float* __restrict__ Wk,
                       const float* __restrict__ Wv, const float* __restrict__ Wo,
                       bf16* __restrict__ Wstack, bf16* __restrict__ Wob) {
  int i = blockIdx.x * blockDim.x + threadIdx.x;
  if (i < 128 * 128) {
    Wstack[i]         = (bf16)Wq[i];
    Wstack[16384 + i] = (bf16)Wk[i];
    Wstack[32768 + i] = (bf16)Wv[i];
    Wob[i]            = (bf16)Wo[i];
  }
}

// ---------------- zero the atomic accumulators (every call) ----------------
__global__ void k_zero(float* __restrict__ ksum, float* __restrict__ ctx,
                       float* __restrict__ gsum, float* __restrict__ gsq) {
  int i = blockIdx.x * blockDim.x + threadIdx.x;
  if (i < B_ * H_ * D_) ksum[i] = 0.f;
  if (i < B_ * H_ * D_ * D_) ctx[i] = 0.f;
  if (i < B_) { gsum[i] = 0.f; gsq[i] = 0.f; }
}

// ---------------- K1: QKV projection + q-softmax + exp(k) + row sums -------
// grid (N/64, B), 384 threads = 12 waves; wave = (matrix m, head h)
__global__ __launch_bounds__(384) void k_proj(
    const float* __restrict__ x, const bf16* __restrict__ Wstack,
    bf16* __restrict__ Qs,      // [B][H][N][32]  (d contiguous)
    bf16* __restrict__ Kexp,    // [B][128][N]
    bf16* __restrict__ Vv,      // [B][128][N]
    float* __restrict__ ksum) { // [B][H][32]
  const int nbase = blockIdx.x * 64;
  const int b = blockIdx.y;
  const int tid = threadIdx.x;

  __shared__ bf16 xT[64 * 136];   // [n][c], padded to dodge bank conflicts
  for (int idx = tid; idx < 128 * 64; idx += 384) {
    int c = idx >> 6, n = idx & 63;
    xT[n * 136 + c] = (bf16)x[((size_t)b * C_ + c) * N_ + nbase + n];
  }
  __syncthreads();

  const int w = tid >> 5, lane = tid & 31;
  const int m = w >> 2, h = w & 3;            // m: 0=Q 1=K 2=V
  const int base_row = m * 128 + h * 32;
  const int half = lane >> 4, nl = lane & 15;

  v8f vz = {0.f, 0.f, 0.f, 0.f, 0.f, 0.f, 0.f, 0.f};
  v8f acc[2][4];
#pragma unroll
  for (int t = 0; t < 2; ++t)
#pragma unroll
    for (int ct = 0; ct < 4; ++ct) acc[t][ct] = vz;

#pragma unroll
  for (int ks = 0; ks < 4; ++ks) {
    v16bf a0 = load_frag_a(Wstack + (size_t)base_row * 128 + ks * 32, 128, lane);
    v16bf a1 = load_frag_a(Wstack + (size_t)(base_row + 16) * 128 + ks * 32, 128, lane);
#pragma unroll
    for (int ct = 0; ct < 4; ++ct) {
      v16bf bb = load_frag_b(xT + ct * 16 * 136 + ks * 32, 136, lane);
      acc[0][ct] = wmma_bf16(a0, bb, acc[0][ct]);
      acc[1][ct] = wmma_bf16(a1, bb, acc[1][ct]);
    }
  }

  if (m == 0) {
    // softmax over d_head (32 rows/column; column lives in lanes l and l^16)
#pragma unroll
    for (int ct = 0; ct < 4; ++ct) {
      float mx = -1e30f;
#pragma unroll
      for (int t = 0; t < 2; ++t)
#pragma unroll
        for (int v = 0; v < 8; ++v) mx = fmaxf(mx, acc[t][ct][v]);
      mx = fmaxf(mx, __shfl_xor(mx, 16, 32));
      float e[2][8], s = 0.f;
#pragma unroll
      for (int t = 0; t < 2; ++t)
#pragma unroll
        for (int v = 0; v < 8; ++v) { e[t][v] = __expf(acc[t][ct][v] - mx); s += e[t][v]; }
      s += __shfl_xor(s, 16, 32);
      float inv = SCALE_ / s;
      size_t nglob = (size_t)nbase + ct * 16 + nl;
      bf16* dst = Qs + (((size_t)b * H_ + h) * N_ + nglob) * 32;
#pragma unroll
      for (int t = 0; t < 2; ++t) {
        int d0 = t * 16 + half * 8;
#pragma unroll
        for (int v = 0; v < 8; ++v) dst[d0 + v] = (bf16)(e[t][v] * inv);
      }
    }
  } else if (m == 1) {
    // exp(k) (values ~N(0,0.57): exp w/o max-sub is safe in fp32) + row sums
    float rsum[2][8];
#pragma unroll
    for (int t = 0; t < 2; ++t)
#pragma unroll
      for (int v = 0; v < 8; ++v) rsum[t][v] = 0.f;
#pragma unroll
    for (int ct = 0; ct < 4; ++ct) {
      size_t nglob = (size_t)nbase + ct * 16 + nl;
#pragma unroll
      for (int t = 0; t < 2; ++t)
#pragma unroll
        for (int v = 0; v < 8; ++v) {
          int d = t * 16 + v + half * 8;
          float ev = __expf(acc[t][ct][v]);
          rsum[t][v] += ev;
          Kexp[((size_t)b * C_ + h * 32 + d) * N_ + nglob] = (bf16)ev;
        }
    }
#pragma unroll
    for (int t = 0; t < 2; ++t)
#pragma unroll
      for (int v = 0; v < 8; ++v) {
        float s = rsum[t][v];
#pragma unroll
        for (int off = 1; off < 16; off <<= 1) s += __shfl_xor(s, off, 32);
        if (nl == 0) {
          int d = t * 16 + v + half * 8;
          atomicAdd(&ksum[((size_t)b * H_ + h) * 32 + d], s);
        }
      }
  } else {
    // V: just store bf16
#pragma unroll
    for (int ct = 0; ct < 4; ++ct) {
      size_t nglob = (size_t)nbase + ct * 16 + nl;
#pragma unroll
      for (int t = 0; t < 2; ++t)
#pragma unroll
        for (int v = 0; v < 8; ++v) {
          int d = t * 16 + v + half * 8;
          Vv[((size_t)b * C_ + h * 32 + d) * N_ + nglob] = (bf16)acc[t][ct][v];
        }
    }
  }
}

// ---------------- K2: ctx[d,e] += sum_n exp(k)[d,n] * v[e,n] ---------------
// grid (32 chunks, H, B), 128 threads = 4 waves; wave = (d-tile, e-tile)
__global__ __launch_bounds__(128) void k_ctx(
    const bf16* __restrict__ Kexp, const bf16* __restrict__ Vv,
    float* __restrict__ ctx) {
  const int chunk = blockIdx.x, h = blockIdx.y, b = blockIdx.z;
  const int w = threadIdx.x >> 5, lane = threadIdx.x & 31;
  const int dt = w >> 1, et = w & 1;
  const int half = lane >> 4, nl = lane & 15;
  const bf16* Abase = Kexp + ((size_t)b * C_ + h * 32 + dt * 16) * N_;
  const bf16* Bbase = Vv   + ((size_t)b * C_ + h * 32 + et * 16) * N_;
  const int k0 = chunk * (N_ / 32);   // 2048-wide contraction chunk

  v8f acc = {0.f, 0.f, 0.f, 0.f, 0.f, 0.f, 0.f, 0.f};
  for (int ks = 0; ks < 2048 / 32; ++ks) {
    int k = k0 + ks * 32;
    __builtin_prefetch(Abase + k + 128, 0, 1);
    __builtin_prefetch(Bbase + k + 128, 0, 1);
    v16bf a  = load_frag_a(Abase + k, N_, lane);
    v16bf bb = load_frag_b(Bbase + k, N_, lane);
    acc = wmma_bf16(a, bb, acc);
  }
#pragma unroll
  for (int v = 0; v < 8; ++v) {
    int d = dt * 16 + v + half * 8;
    int e = et * 16 + nl;
    atomicAdd(&ctx[(((size_t)b * H_ + h) * 32 + d) * 32 + e], acc[v]);
  }
}

// ---------------- K3: ctxT[e,d] = bf16(ctx[d,e] / ksum[d]) -----------------
__global__ void k_ctx_fin(const float* __restrict__ ctx,
                          const float* __restrict__ ksum,
                          bf16* __restrict__ ctxT) {
  int idx = blockIdx.x * blockDim.x + threadIdx.x;
  if (idx >= B_ * H_ * D_ * D_) return;
  int e = idx & 31, d = (idx >> 5) & 31, h = (idx >> 10) & 3, b = idx >> 12;
  float val = ctx[(((size_t)b * H_ + h) * 32 + d) * 32 + e] /
              ksum[((size_t)b * H_ + h) * 32 + d];
  ctxT[(((size_t)b * H_ + h) * 32 + e) * 32 + d] = (bf16)val;
}

// ---------------- K4: att = ctxT*Q ; o = Wo*att + bo ; GN partial sums -----
// grid (N/64, B), 256 threads = 8 waves
__global__ __launch_bounds__(256) void k_out(
    const bf16* __restrict__ Qs, const bf16* __restrict__ ctxT,
    const bf16* __restrict__ Wob, const float* __restrict__ bo,
    float* __restrict__ o, float* __restrict__ gsum, float* __restrict__ gsq) {
  const int nbase = blockIdx.x * 64;
  const int b = blockIdx.y;
  const int tid = threadIdx.x, w = tid >> 5, lane = tid & 31;
  const int half = lane >> 4, nl = lane & 15;

  __shared__ bf16 attS[64 * 136];   // [n][hid]
  __shared__ float red[16];

  // stage 1: wave -> (head h, e-tile et): att rows = h*32+et*16 .. +15
  {
    const int h = w >> 1, et = w & 1;
    v16bf a = load_frag_a(ctxT + ((size_t)b * H_ + h) * 1024 + et * 16 * 32, 32, lane);
#pragma unroll
    for (int ct = 0; ct < 4; ++ct) {
      v16bf bb = load_frag_b(Qs + (((size_t)b * H_ + h) * N_ + nbase + ct * 16) * 32,
                             32, lane);
      v8f acc = {0.f, 0.f, 0.f, 0.f, 0.f, 0.f, 0.f, 0.f};
      acc = wmma_bf16(a, bb, acc);
      int n = ct * 16 + nl;
      int hid0 = h * 32 + et * 16 + half * 8;
#pragma unroll
      for (int v = 0; v < 8; ++v) attS[n * 136 + hid0 + v] = (bf16)acc[v];
    }
  }
  __syncthreads();

  // stage 2: wave -> output row tile w (rows 16w..16w+15)
  float s1 = 0.f, s2 = 0.f;
  {
    v8f vz = {0.f, 0.f, 0.f, 0.f, 0.f, 0.f, 0.f, 0.f};
    v8f acc[4] = {vz, vz, vz, vz};
#pragma unroll
    for (int ks = 0; ks < 4; ++ks) {
      v16bf a = load_frag_a(Wob + (size_t)(w * 16) * 128 + ks * 32, 128, lane);
#pragma unroll
      for (int ct = 0; ct < 4; ++ct) {
        v16bf bb = load_frag_b(attS + ct * 16 * 136 + ks * 32, 136, lane);
        acc[ct] = wmma_bf16(a, bb, acc[ct]);
      }
    }
#pragma unroll
    for (int ct = 0; ct < 4; ++ct) {
      size_t n = (size_t)nbase + ct * 16 + nl;
#pragma unroll
      for (int v = 0; v < 8; ++v) {
        int row = w * 16 + v + half * 8;
        float val = acc[ct][v] + bo[row];
        o[((size_t)b * C_ + row) * N_ + n] = val;
        s1 += val;
        s2 += val * val;
      }
    }
  }
#pragma unroll
  for (int off = 1; off < 32; off <<= 1) {
    s1 += __shfl_xor(s1, off, 32);
    s2 += __shfl_xor(s2, off, 32);
  }
  if (lane == 0) { red[w] = s1; red[8 + w] = s2; }
  __syncthreads();
  if (tid == 0) {
    float a = 0.f, c2 = 0.f;
    for (int i = 0; i < 8; ++i) { a += red[i]; c2 += red[8 + i]; }
    atomicAdd(&gsum[b], a);
    atomicAdd(&gsq[b], c2);
  }
}

// ---------------- K5: finalize GroupNorm stats -----------------------------
__global__ void k_stats(const float* __restrict__ gsum, const float* __restrict__ gsq,
                        float* __restrict__ meanrs) {
  int b = threadIdx.x;
  if (b >= B_) return;
  const float cnt = (float)C_ * (float)N_;
  float mean = gsum[b] / cnt;
  float var = gsq[b] / cnt - mean * mean;
  meanrs[b] = mean;
  meanrs[B_ + b] = rsqrtf(var + EPS_);
}

// ---------------- K6: normalize + affine + residual ------------------------
__global__ __launch_bounds__(256) void k_norm(
    const float* __restrict__ o, const float* __restrict__ x,
    const float* __restrict__ gamma, const float* __restrict__ beta,
    const float* __restrict__ meanrs, float* __restrict__ out) {
  const size_t total = (size_t)B_ * C_ * N_ / 4;
  for (size_t i = (size_t)blockIdx.x * blockDim.x + threadIdx.x; i < total;
       i += (size_t)gridDim.x * blockDim.x) {
    size_t flat = i * 4;
    int c = (int)((flat >> 16) & 127);   // N_ = 2^16
    int b = (int)(flat >> 23);           // C_*N_ = 2^23
    float4 ov = ((const float4*)o)[i];
    float4 xv = ((const float4*)x)[i];
    float a = meanrs[B_ + b] * gamma[c];
    float sh = beta[c] - meanrs[b] * a;
    float4 r;
    r.x = ov.x * a + sh + xv.x;
    r.y = ov.y * a + sh + xv.y;
    r.z = ov.z * a + sh + xv.z;
    r.w = ov.w * a + sh + xv.w;
    ((float4*)out)[i] = r;
  }
}

// ---------------- workspace layout -----------------------------------------
static const size_t OFF_QS = 0;                       // 128 MB bf16 [B][H][N][32]
static const size_t OFF_KE = 128ull << 20;            // 128 MB bf16
static const size_t OFF_V  = 256ull << 20;            // 128 MB bf16
static const size_t OFF_O  = 128ull << 20;            // 256 MB f32 (aliases KE+V, used after k_ctx)
static const size_t OFF_SM = 384ull << 20;
static const size_t OFF_KSUM   = OFF_SM;              // 4 KB
static const size_t OFF_CTX    = OFF_SM + (1ull << 20);
static const size_t OFF_CTXT   = OFF_SM + (2ull << 20);
static const size_t OFF_GSUM   = OFF_SM + (3ull << 20);
static const size_t OFF_GSQ    = OFF_GSUM + 4096;
static const size_t OFF_MEANRS = OFF_GSUM + 8192;
static const size_t OFF_WSTK   = OFF_SM + (4ull << 20); // 96 KB bf16
static const size_t OFF_WOB    = OFF_SM + (5ull << 20); // 32 KB bf16

extern "C" void kernel_launch(void* const* d_in, const int* in_sizes, int n_in,
                              void* d_out, int out_size, void* d_ws, size_t ws_size,
                              hipStream_t stream) {
  const float* x     = (const float*)d_in[0];
  const float* Wq    = (const float*)d_in[1];
  const float* Wk    = (const float*)d_in[2];
  const float* Wv    = (const float*)d_in[3];
  const float* Wo    = (const float*)d_in[4];
  const float* bo    = (const float*)d_in[5];
  const float* gamma = (const float*)d_in[6];
  const float* beta  = (const float*)d_in[7];
  float* out = (float*)d_out;

  char* ws = (char*)d_ws;
  bf16*  Qs     = (bf16*)(ws + OFF_QS);
  bf16*  Kexp   = (bf16*)(ws + OFF_KE);
  bf16*  Vv     = (bf16*)(ws + OFF_V);
  float* o      = (float*)(ws + OFF_O);
  float* ksum   = (float*)(ws + OFF_KSUM);
  float* ctx    = (float*)(ws + OFF_CTX);
  bf16*  ctxT   = (bf16*)(ws + OFF_CTXT);
  float* gsum   = (float*)(ws + OFF_GSUM);
  float* gsq    = (float*)(ws + OFF_GSQ);
  float* meanrs = (float*)(ws + OFF_MEANRS);
  bf16*  Wstk   = (bf16*)(ws + OFF_WSTK);
  bf16*  Wob    = (bf16*)(ws + OFF_WOB);

  k_pack<<<64, 256, 0, stream>>>(Wq, Wk, Wv, Wo, Wstk, Wob);
  k_zero<<<128, 256, 0, stream>>>(ksum, ctx, gsum, gsq);
  k_proj<<<dim3(N_ / 64, B_), 384, 0, stream>>>(x, Wstk, Qs, Kexp, Vv, ksum);
  k_ctx<<<dim3(32, H_, B_), 128, 0, stream>>>(Kexp, Vv, ctx);
  k_ctx_fin<<<128, 256, 0, stream>>>(ctx, ksum, ctxT);
  k_out<<<dim3(N_ / 64, B_), 256, 0, stream>>>(Qs, ctxT, Wob, bo, o, gsum, gsq);
  k_stats<<<1, 32, 0, stream>>>(gsum, gsq, meanrs);
  k_norm<<<4096, 256, 0, stream>>>(o, x, gamma, beta, meanrs, out);
}
